// NKAT_ThinkingBlock_19859928777062
// MI455X (gfx1250) — matrix operand order repaired and checked by reading, not compiled, not run on hardware
//
#include <hip/hip_runtime.h>
#include <hip/hip_bf16.h>
#include <stdint.h>

#define DIM 2048
#define NB  256
#define TOK 16384

typedef __attribute__((ext_vector_type(16))) __bf16 v16bf;
typedef __attribute__((ext_vector_type(8)))  float  v8f;

// ---------------------------------------------------------------------------
// Kernel 1: Rg[k] = sigmoid(decay[k]) (row-scaled) * expm(0.5*(A - A^T))
// ---------------------------------------------------------------------------
__global__ __launch_bounds__(64)
void expm_kernel(const float* __restrict__ so8,
                 const float* __restrict__ decay,
                 float* __restrict__ Rg) {
  __shared__ float sA[64];
  __shared__ float sM[64];
  __shared__ int   s_s;
  const int k = blockIdx.x;
  const int t = threadIdx.x;
  const int j = t >> 3, i = t & 7;

  const float raw  = so8[k * 64 + j * 8 + i];
  const float rawT = so8[k * 64 + i * 8 + j];
  float a = 0.5f * (raw - rawT);
  sA[t] = a;
  __syncthreads();

  if (t == 0) {  // inf-norm -> scaling count
    float mx = 0.f;
    for (int r = 0; r < 8; ++r) {
      float s = 0.f;
      for (int c = 0; c < 8; ++c) s += fabsf(sA[r * 8 + c]);
      mx = fmaxf(mx, s);
    }
    int sc = 0;
    while (mx > 0.25f && sc < 24) { mx *= 0.5f; ++sc; }
    s_s = sc;
  }
  __syncthreads();
  const int sc = s_s;
  a = ldexpf(a, -sc);
  sA[t] = a;
  sM[t] = (j == i) ? 1.f : 0.f;
  __syncthreads();

  for (int n = 12; n >= 1; --n) {  // Horner: M = I + (A/n)*M
    float acc = 0.f;
#pragma unroll
    for (int m = 0; m < 8; ++m) acc += sA[j * 8 + m] * sM[m * 8 + i];
    acc = acc / (float)n + ((j == i) ? 1.f : 0.f);
    __syncthreads();
    sM[t] = acc;
    __syncthreads();
  }
  for (int q = 0; q < sc; ++q) {  // squaring
    float acc = 0.f;
#pragma unroll
    for (int m = 0; m < 8; ++m) acc += sM[j * 8 + m] * sM[m * 8 + i];
    __syncthreads();
    sM[t] = acc;
    __syncthreads();
  }
  const float g = 1.f / (1.f + expf(-decay[k * 8 + j]));
  Rg[k * 64 + j * 8 + i] = g * sM[t];
}

// ---------------------------------------------------------------------------
// Kernel 2: W'[r,c] = sum_j W[r,(c/8)*8+j] * Rg[c/8][j][c%8]  -> bf16 [N][K]
// ---------------------------------------------------------------------------
__global__ __launch_bounds__(256)
void buildB_kernel(const float* __restrict__ W,
                   const float* __restrict__ Rg,
                   __bf16* __restrict__ Bw) {
  __shared__ float sW[256];
  const int r  = blockIdx.y;
  const int c0 = blockIdx.x * 256;
  const int t  = threadIdx.x;
  sW[t] = W[(size_t)r * DIM + c0 + t];
  __syncthreads();
  const int c  = c0 + t;
  const int k  = c >> 3, i = c & 7;
  const int lb = t & ~7;
  float sum = 0.f;
#pragma unroll
  for (int jj = 0; jj < 8; ++jj)
    sum += sW[lb + jj] * Rg[k * 64 + jj * 8 + i];
  Bw[(size_t)r * DIM + c] = (__bf16)sum;
}

// ---------------------------------------------------------------------------
// Kernel 3: Y = X + relu(X @ W'^T + b)^2  — WMMA bf16 GEMM, double-buffered
// LDS, async-to-LDS B tiles, fused epilogue.
// ---------------------------------------------------------------------------
#define BM 128
#define BN 128
#define BK 32
#define NT (DIM / BK)
#define LSTR 40  // bf16 elems per LDS row (80B: 16B-aligned, bank-spread)

__global__ __launch_bounds__(256)
void gemm_kernel(const float* __restrict__ X,
                 const __bf16* __restrict__ Bw,
                 const float* __restrict__ bias,
                 float* __restrict__ Y) {
  __shared__ __bf16 sA[2][BM * LSTR];
  __shared__ __bf16 sB[2][BN * LSTR];

  const int tid  = threadIdx.x;
  const int wid  = tid >> 5;
  const int lane = tid & 31;
  const int m_base = blockIdx.y * BM;
  const int n_base = blockIdx.x * BN;
  const int wm0 = (wid & 3) * 32;
  const int wn0 = (wid >> 2) * 64;
  const int half = lane >> 4;
  const int lr   = lane & 15;

  // per-thread staging coordinates
  const int arow = tid >> 3,  ac4 = (tid & 7) * 4;     // +row stride 32 per it
  const int brow = tid >> 2,  bc8 = (tid & 3) * 8;     // +row stride 64 per it

  v8f acc[2][4];
#pragma unroll
  for (int mf = 0; mf < 2; ++mf)
#pragma unroll
    for (int nf = 0; nf < 4; ++nf) acc[mf][nf] = (v8f){0,0,0,0,0,0,0,0};

  // ---- helpers -----------------------------------------------------------
  auto loadA_g = [&](int kt, float4* f) {
    const int k0 = kt * BK;
#pragma unroll
    for (int it = 0; it < 4; ++it)
      f[it] = *(const float4*)(X + (size_t)(m_base + arow + it * 32) * DIM + k0 + ac4);
  };
  auto storeA_s = [&](int buf, const float4* f) {
#pragma unroll
    for (int it = 0; it < 4; ++it) {
      union { __bf16 h[4]; uint2 u2; } p;
      p.h[0] = (__bf16)f[it].x; p.h[1] = (__bf16)f[it].y;
      p.h[2] = (__bf16)f[it].z; p.h[3] = (__bf16)f[it].w;
      *(uint2*)&sA[buf][(arow + it * 32) * LSTR + ac4] = p.u2;
    }
  };
  auto loadB_async = [&](int kt, int buf) {
    const int k0 = kt * BK;
#pragma unroll
    for (int it = 0; it < 2; ++it) {
      unsigned ldsa = (unsigned)(uintptr_t)&sB[buf][(brow + it * 64) * LSTR + bc8];
      const __bf16* gp = Bw + (size_t)(n_base + brow + it * 64) * DIM + k0 + bc8;
      asm volatile("global_load_async_to_lds_b128 %0, %1, off"
                   :: "v"(ldsa), "v"(gp) : "memory");
    }
  };

  // ---- preload tile 0 ----------------------------------------------------
  {
    float4 f0[4];
    loadA_g(0, f0);
    loadB_async(0, 0);
    storeA_s(0, f0);
  }
  asm volatile("s_wait_asynccnt 0x0" ::: "memory");
  __syncthreads();

  // ---- main K loop -------------------------------------------------------
  float4 fnext[4];
  for (int kt = 0; kt < NT; ++kt) {
    const int cur = kt & 1;
    const int nxt = cur ^ 1;
    const bool has_next = (kt + 1 < NT);

    if (has_next) {
      loadA_g(kt + 1, fnext);      // global -> regs (hidden under WMMA)
      loadB_async(kt + 1, nxt);    // global -> LDS, ASYNCcnt
      __builtin_prefetch(X + (size_t)(m_base + arow) * DIM + (kt + 2) * BK, 0, 3);
    }

    // fragments (ISA 16-bit A 16x32: lanes0-15 K{0..7,16..23}, lanes16-31
    // K{8..15,24..31}; B 32x16: same K order as A)
    v16bf afrag[2], bfrag[4];
#pragma unroll
    for (int mf = 0; mf < 2; ++mf) {
      const int rrow = wm0 + mf * 16 + lr;
      union { v16bf v; uint4 q[2]; } u;
      u.q[0] = *(const uint4*)&sA[cur][rrow * LSTR + half * 8];
      u.q[1] = *(const uint4*)&sA[cur][rrow * LSTR + 16 + half * 8];
      afrag[mf] = u.v;
    }
#pragma unroll
    for (int nf = 0; nf < 4; ++nf) {
      const int rrow = wn0 + nf * 16 + lr;
      union { v16bf v; uint4 q[2]; } u;
      u.q[0] = *(const uint4*)&sB[cur][rrow * LSTR + half * 8];
      u.q[1] = *(const uint4*)&sB[cur][rrow * LSTR + 16 + half * 8];
      bfrag[nf] = u.v;
    }
#pragma unroll
    for (int mf = 0; mf < 2; ++mf)
#pragma unroll
      for (int nf = 0; nf < 4; ++nf)
        acc[mf][nf] = __builtin_amdgcn_wmma_f32_16x16x32_bf16(
            false, afrag[mf], false, bfrag[nf], (short)0, acc[mf][nf],
            false, false);

    if (has_next) storeA_s(nxt, fnext);  // regs -> LDS (other buffer)

    asm volatile("s_wait_asynccnt 0x0" ::: "memory");
    __syncthreads();
  }

  // ---- fused epilogue: bias, relu^2, +residual ---------------------------
#pragma unroll
  for (int nf = 0; nf < 4; ++nf) {
    const int gr = n_base + wn0 + nf * 16 + lr;
    const float bv = bias[gr];
#pragma unroll
    for (int mf = 0; mf < 2; ++mf) {
      const int tbase = m_base + wm0 + mf * 16 + half * 8;
#pragma unroll
      for (int v = 0; v < 8; ++v) {
        const int tt = tbase + v;
        float val = acc[mf][nf][v] + bv;
        val = fmaxf(val, 0.f);
        val = val * val;
        Y[(size_t)tt * DIM + gr] = X[(size_t)tt * DIM + gr] + val;
      }
    }
  }
}

// ---------------------------------------------------------------------------
// Kernel 4: in-place LayerNorm over last dim (one block per row of 2048).
// ---------------------------------------------------------------------------
__global__ __launch_bounds__(256)
void ln_kernel(float* __restrict__ Y,
               const float* __restrict__ lw,
               const float* __restrict__ lb) {
  __shared__ float s_sum[256];
  __shared__ float s_sq[256];
  const size_t row = blockIdx.x;
  const int tid = threadIdx.x;
  float* p = Y + row * DIM + tid * 8;
  const float4 v0 = *(const float4*)(p);
  const float4 v1 = *(const float4*)(p + 4);
  float sum = v0.x + v0.y + v0.z + v0.w + v1.x + v1.y + v1.z + v1.w;
  float sq  = v0.x*v0.x + v0.y*v0.y + v0.z*v0.z + v0.w*v0.w +
              v1.x*v1.x + v1.y*v1.y + v1.z*v1.z + v1.w*v1.w;
  s_sum[tid] = sum; s_sq[tid] = sq;
  __syncthreads();
  for (int off = 128; off > 0; off >>= 1) {
    if (tid < off) { s_sum[tid] += s_sum[tid + off]; s_sq[tid] += s_sq[tid + off]; }
    __syncthreads();
  }
  const float mu   = s_sum[0] * (1.f / DIM);
  const float var  = s_sq[0] * (1.f / DIM) - mu * mu;
  const float rstd = rsqrtf(var + 1e-5f);
  const float4 w0 = *(const float4*)(lw + tid * 8);
  const float4 w1 = *(const float4*)(lw + tid * 8 + 4);
  const float4 b0 = *(const float4*)(lb + tid * 8);
  const float4 b1 = *(const float4*)(lb + tid * 8 + 4);
  float4 o0, o1;
  o0.x = (v0.x - mu) * rstd * w0.x + b0.x;
  o0.y = (v0.y - mu) * rstd * w0.y + b0.y;
  o0.z = (v0.z - mu) * rstd * w0.z + b0.z;
  o0.w = (v0.w - mu) * rstd * w0.w + b0.w;
  o1.x = (v1.x - mu) * rstd * w1.x + b1.x;
  o1.y = (v1.y - mu) * rstd * w1.y + b1.y;
  o1.z = (v1.z - mu) * rstd * w1.z + b1.z;
  o1.w = (v1.w - mu) * rstd * w1.w + b1.w;
  *(float4*)(p)     = o0;
  *(float4*)(p + 4) = o1;
}

// ---------------------------------------------------------------------------
extern "C" void kernel_launch(void* const* d_in, const int* in_sizes, int n_in,
                              void* d_out, int out_size, void* d_ws, size_t ws_size,
                              hipStream_t stream) {
  const float* x     = (const float*)d_in[0];
  const float* so8   = (const float*)d_in[1];
  const float* decay = (const float*)d_in[2];
  const float* W     = (const float*)d_in[3];
  const float* bmix  = (const float*)d_in[4];
  const float* lnw   = (const float*)d_in[5];
  const float* lnb   = (const float*)d_in[6];
  float* out = (float*)d_out;

  float*  Rg = (float*)d_ws;                        // 64 KB
  __bf16* Bw = (__bf16*)((char*)d_ws + 65536);      // 8 MB

  expm_kernel<<<NB, 64, 0, stream>>>(so8, decay, Rg);
  buildB_kernel<<<dim3(DIM / 256, DIM), 256, 0, stream>>>(W, Rg, Bw);
  gemm_kernel<<<dim3(DIM / BN, TOK / BM), 256, 0, stream>>>(x, Bw, bmix, out);
  ln_kernel<<<TOK, 256, 0, stream>>>(out, lnw, lnb);
}
